// HarmonicWaveGenerator_56118042689690
// MI455X (gfx1250) — matrix operand order repaired and checked by reading, not compiled, not run on hardware
//
#include <hip/hip_runtime.h>
#include <math.h>

typedef __attribute__((ext_vector_type(16))) _Float16 v16h;
typedef __attribute__((ext_vector_type(8)))  _Float16 v8h;
typedef __attribute__((ext_vector_type(8)))  float    v8f;
typedef __attribute__((ext_vector_type(4)))  unsigned int v4u;
typedef __attribute__((ext_vector_type(8)))  int      v8i;
typedef __attribute__((ext_vector_type(4)))  int      v4i;

static constexpr int   TF   = 400;
static constexpr int   BB   = 8;
static constexpr int   HH   = 32;
static constexpr int   HOPN = 240;
static constexpr float SRF  = 22050.0f;
static constexpr float PI_F  = 3.14159265358979323846f;
static constexpr float TPI_F = 6.28318530717958647692f;

// ---------------------------------------------------------------------------
// Prep: fold BN into weights, convert to f16, pad K to 32 with zeros
// ---------------------------------------------------------------------------
__global__ void fold_weights(const float* __restrict__ w, const float* __restrict__ bsrc,
                             const float* __restrict__ g, const float* __restrict__ be,
                             const float* __restrict__ m, const float* __restrict__ v,
                             _Float16* __restrict__ w16, float* __restrict__ bfold,
                             int OC, int ICK, int Kpad)
{
    const float eps = 1e-5f;
    int stride = gridDim.x * blockDim.x;
    int total = OC * Kpad;
    for (int idx = blockIdx.x * blockDim.x + threadIdx.x; idx < total; idx += stride) {
        int oc = idx / Kpad;
        int k  = idx - oc * Kpad;
        float val = 0.0f;
        if (k < ICK) {
            float s = g ? g[oc] * rsqrtf(v[oc] + eps) : 1.0f;
            val = w[oc * ICK + k] * s;
        }
        w16[idx] = (_Float16)val;
    }
    for (int oc = blockIdx.x * blockDim.x + threadIdx.x; oc < OC; oc += stride) {
        float s = g ? g[oc] * rsqrtf(v[oc] + eps) : 1.0f;
        bfold[oc] = g ? (bsrc[oc] - m[oc]) * s + be[oc] : bsrc[oc];
    }
}

// mel (f32) -> rows 0..79 of comb buffer [B][96][400] (f16)
__global__ void mel_to_comb(const float* __restrict__ mel, _Float16* __restrict__ comb)
{
    int stride = gridDim.x * blockDim.x;
    const int total = BB * 80 * TF;
    for (int idx = blockIdx.x * blockDim.x + threadIdx.x; idx < total; idx += stride) {
        int b = idx / (80 * TF);
        int r = idx % (80 * TF);
        comb[(size_t)b * 96 * TF + r] = (_Float16)mel[idx];
    }
}

// f0 -> f0_hz (needs global mean first); also zero-init wmax slots
__global__ void f0_process(const float* __restrict__ f0, const float* __restrict__ voicing,
                           float* __restrict__ f0_hz, unsigned* __restrict__ wmaxBits)
{
    __shared__ float red[256];
    int tid = threadIdx.x;
    float s = 0.0f;
    for (int i = tid; i < BB * TF; i += 256) s += f0[i];
    red[tid] = s; __syncthreads();
    for (int w = 128; w > 0; w >>= 1) { if (tid < w) red[tid] += red[tid + w]; __syncthreads(); }
    float mean = red[0] / (float)(BB * TF);
    for (int i = tid; i < BB * TF; i += 256) {
        float x = f0[i];
        float hz;
        if (mean < 0.0f)       hz = __expf(x);
        else if (mean < 50.0f) hz = 440.0f * exp2f((x - 69.0f) / 12.0f);
        else                   hz = x;
        hz = fminf(fmaxf(hz, 50.0f), 4000.0f);
        float vd = voicing[i];
        f0_hz[i] = hz * vd + 100.0f * (1.0f - vd);
    }
    if (tid < BB) wmaxBits[tid] = 0u;
}

// tiny f0 branch convs (1->16->16), result into comb rows 80..95 (f16, relu)
__global__ void f0_branch(const float* __restrict__ f0,
                          const float* __restrict__ w1, const float* __restrict__ b1,
                          const float* __restrict__ w2, const float* __restrict__ b2,
                          _Float16* __restrict__ comb)
{
    int b = blockIdx.x;
    int tid = threadIdx.x;
    __shared__ float h1[16 * 402];           // padded [-1..400]
    const float* fb = f0 + b * TF;
    for (int idx = tid; idx < 16 * 402; idx += 256) {
        int ic = idx / 402;
        int tt = idx % 402 - 1;
        float acc = 0.0f;
        if (tt >= 0 && tt < TF) {
            acc = b1[ic];
            for (int k = 0; k < 3; k++) {
                int t = tt + k - 1;
                if (t >= 0 && t < TF) acc += w1[ic * 3 + k] * fb[t];
            }
            acc = fmaxf(acc, 0.0f);
        }
        h1[idx] = acc;
    }
    __syncthreads();
    for (int idx = tid; idx < 16 * TF; idx += 256) {
        int oc = idx / TF, t = idx % TF;
        float acc = b2[oc];
        for (int ic = 0; ic < 16; ic++)
            for (int k = 0; k < 3; k++)
                acc += w2[(oc * 16 + ic) * 3 + k] * h1[ic * 402 + t + k];
        acc = fmaxf(acc, 0.0f);
        comb[((size_t)b * 96 + 80 + oc) * TF + t] = (_Float16)acc;
    }
}

// ---------------------------------------------------------------------------
// WMMA conv1d (k=3, pad=1) as im2col GEMM.
// X tile (IC rows x 18 cols, row stride 400) staged into LDS by the Tensor
// Data Mover (one tensor_load_to_lds per block, TENSORcnt-tracked), halo
// columns zeroed at grid edges. Block = WPB waves sharing one LDS-transposed
// im2col tile; each wave owns one 16-row M-tile. Fully unrolled K loop:
//   A fragment = 2x contiguous global b128 loads (zero-padded weights)
//   B fragment = 1x contiguous 32B LDS load (2x ds_load_b128)
// MODE 0: relu->f16   MODE 1: sigmoid*voiced+1e-8->f32   MODE 2: tanh->f32
// ---------------------------------------------------------------------------
template <int IC, int OC, int MODE>
__global__ __launch_bounds__(OC >= 64 ? 128 : 64)
void conv_wmma(const _Float16* __restrict__ X,   // [B][CSTRIDE][400] f16
               const _Float16* __restrict__ W16, // [OC][Kpad] f16 (BN folded, 0-padded)
               const float* __restrict__ bias,   // [OC] f32 (BN folded)
               _Float16* __restrict__ outH,      // MODE 0
               float* __restrict__ outF,         // MODE 1/2
               const float* __restrict__ voiced, // [B][400] (MODE 1)
               int CSTRIDE)
{
    constexpr int Ktot = IC * 3;
    constexpr int Kpad = ((Ktot + 31) / 32) * 32;
    constexpr int WPB  = OC >= 64 ? 4 : 2;       // waves per block
    constexpr int NTH  = WPB * 32;

    const int t0   = blockIdx.y * 16;
    const int b    = blockIdx.z;
    const int tid  = threadIdx.x;
    const int lane = tid & 31;
    const int wv   = tid >> 5;
    const int row  = lane & 15;
    const int hi   = lane >> 4;                  // half-wave K split
    const int m0   = blockIdx.x * (16 * WPB) + wv * 16;

    __shared__ __align__(16) _Float16 xs[IC * 18];     // X tile + halo
    __shared__ __align__(16) _Float16 im[16 * Kpad];   // im2col, transposed [n][K]

    const _Float16* xb = X + (size_t)b * CSTRIDE * TF;

#if __has_builtin(__builtin_amdgcn_tensor_load_to_lds)
    // --- Tensor Data Mover: DMA the IC x 18 tile (row stride 400) into LDS ---
    if (tid < 32) {
        unsigned ldsAddr = (unsigned)(size_t)(&xs[0]);
        unsigned long long ga = (unsigned long long)(size_t)(xb + (t0 - 1));
        v4u g0;
        g0.x = 1u;                                       // count=1, user D#
        g0.y = ldsAddr;                                  // lds_addr
        g0.z = (unsigned)(ga & 0xFFFFFFFFu);             // global_addr[31:0]
        g0.w = (unsigned)((ga >> 32) & 0x01FFFFFFu)      // global_addr[56:32]
             | (2u << 30);                               // type=2 (image)
        v8i g1;
        g1[0] = (int)(1u << 16);                         // data_size=2B
        g1[1] = (int)(400u << 16);                       // tensor_dim0 = 400
        g1[2] = (int)((unsigned)IC << 16);               // tensor_dim1 = IC
        g1[3] = (int)(18u << 16);                        // tile_dim0 = 18
        g1[4] = (int)(unsigned)IC;                       // tile_dim1 = IC
        g1[5] = 400;                                     // tensor_dim0_stride = 400
        g1[6] = 0;                                       // dim1_stride = 0
        g1[7] = 0;
        v4i g2 = {0, 0, 0, 0};
        v4i g3 = {0, 0, 0, 0};
#if defined(__clang_major__) && (__clang_major__ >= 23)
        v8i g4 = {0, 0, 0, 0, 0, 0, 0, 0};
        __builtin_amdgcn_tensor_load_to_lds(g0, g1, g2, g3, g4, 0);
#else
        __builtin_amdgcn_tensor_load_to_lds(g0, g1, g2, g3, 0);
#endif
        __builtin_amdgcn_s_wait_tensorcnt(0);
    }
    __syncthreads();
    // zero halo columns at tensor edges (avoid garbage/NaN into WMMA)
    if (t0 == 0)        for (int ic = tid; ic < IC; ic += NTH) xs[ic * 18 + 0]  = (_Float16)0.0f;
    if (t0 + 16 >= TF)  for (int ic = tid; ic < IC; ic += NTH) xs[ic * 18 + 17] = (_Float16)0.0f;
    __syncthreads();
#else
    for (int idx = tid; idx < IC * 18; idx += NTH) {
        int ic = idx / 18, c = idx % 18;
        int t = t0 + c - 1;
        xs[idx] = (t >= 0 && t < TF) ? xb[ic * TF + t] : (_Float16)0.0f;
    }
    __syncthreads();
#endif

    for (int idx = tid; idx < 16 * Kpad; idx += NTH) {
        int n = idx / Kpad, K = idx - n * Kpad;
        _Float16 val = (_Float16)0.0f;
        if (K < Ktot) { int ic = K / 3, kk = K - ic * 3; val = xs[ic * 18 + n + kk]; }
        im[idx] = val;
    }
    __syncthreads();

    const _Float16* wrow = W16 + (size_t)(m0 + row) * Kpad;
    const _Float16* brow = im + row * Kpad;
    v8f acc = {};
    #pragma unroll
    for (int k0 = 0; k0 < Kpad; k0 += 32) {
        v8h alo = *(const v8h*)(wrow + k0 + hi * 16);       // K = k0+hi*16 .. +7
        v8h ahi = *(const v8h*)(wrow + k0 + hi * 16 + 8);   // K = k0+hi*16+8 .. +15
        v16h afrag = __builtin_shufflevector(alo, ahi,
                       0, 1, 2, 3, 4, 5, 6, 7, 8, 9, 10, 11, 12, 13, 14, 15);
        v16h bfrag = *(const v16h*)(brow + k0 + hi * 16);   // K = k0+hi*16 .. +15
        acc = __builtin_amdgcn_wmma_f32_16x16x32_f16(false, afrag, false, bfrag,
                                                     (short)0, acc, false, false);
    }

    // epilogue: C/D layout -> oc = m0 + i + 8*hi, t = t0 + row
    const int t = t0 + row;
    #pragma unroll
    for (int i = 0; i < 8; i++) {
        int oc = m0 + i + hi * 8;
        float val = acc[i] + bias[oc];
        if constexpr (MODE == 0) {
            val = fmaxf(val, 0.0f);
            outH[((size_t)b * OC + oc) * TF + t] = (_Float16)val;
        } else if constexpr (MODE == 1) {
            float sg = 1.0f / (1.0f + __expf(-val));
            outF[((size_t)b * OC + oc) * TF + t] = sg * voiced[b * TF + t] + 1e-8f;
        } else {
            outF[((size_t)b * OC + oc) * TF + t] = tanhf(val);
        }
    }
}

// ---------------------------------------------------------------------------
// Synthesis: HOP-constant increment -> 400-frame scan, closed form per sample
// ---------------------------------------------------------------------------
__global__ void synth_scan(const float* __restrict__ f0_hz, const float* __restrict__ phases,
                           float* __restrict__ base, float* __restrict__ inc)
{
    int b = blockIdx.x;
    int h = threadIdx.x;                 // 32 harmonics
    float hn = (float)(h + 1);
    float run = 0.0f;
    const float w = TPI_F / SRF;
    for (int f = 0; f < TF; f++) {
        float ic = f0_hz[b * TF + f] * hn * w;
        ic = fminf(fmaxf(ic, -PI_F), PI_F);
        size_t id = ((size_t)b * HH + h) * TF + f;
        inc[id]  = ic;
        base[id] = fmodf(run + phases[id] * PI_F, TPI_F);
        run = fmodf(run + (float)HOPN * ic, TPI_F);
    }
}

__global__ void synth_wave(const float* __restrict__ base, const float* __restrict__ inc,
                           const float* __restrict__ amps,
                           float* __restrict__ wave, unsigned* __restrict__ wmaxBits)
{
    int b  = blockIdx.x;
    int fc = blockIdx.y;                 // 16-frame chunk (25 chunks)
    int tid = threadIdx.x;               // 256
    __shared__ float sb[HH * 16], si[HH * 16], sa[HH * 16];
    __shared__ float red[256];
    int fbase = fc * 16;
    for (int idx = tid; idx < HH * 16; idx += 256) {
        int h = idx >> 4, f = idx & 15;
        size_t id = ((size_t)b * HH + h) * TF + fbase + f;
        sb[idx] = base[id]; si[idx] = inc[id]; sa[idx] = amps[id];
    }
    __syncthreads();
    float lmax = 0.0f;
    for (int s = tid; s < 16 * HOPN; s += 256) {
        int f = s / HOPN, n = s % HOPN;
        float np1 = (float)(n + 1);
        float acc = 0.0f;
        #pragma unroll 4
        for (int h = 0; h < HH; h++) {
            int id = h * 16 + f;
            float p = fmodf(sb[id] + np1 * si[id], TPI_F);
            acc += sa[id] * __sinf(p);
        }
        wave[(size_t)b * (TF * HOPN) + fbase * HOPN + s] = acc;
        lmax = fmaxf(lmax, fabsf(acc));
    }
    red[tid] = lmax; __syncthreads();
    for (int w = 128; w > 0; w >>= 1) { if (tid < w) red[tid] = fmaxf(red[tid], red[tid + w]); __syncthreads(); }
    if (tid == 0) atomicMax(&wmaxBits[b], __float_as_uint(red[0]));
}

__global__ void normalize_out(float* __restrict__ wave, const unsigned* __restrict__ wmaxBits)
{
    int stride = gridDim.x * blockDim.x;
    const int total = BB * TF * HOPN;
    for (int idx = blockIdx.x * blockDim.x + threadIdx.x; idx < total; idx += stride) {
        int b = idx / (TF * HOPN);
        float wm = fmaxf(__uint_as_float(wmaxBits[b]), 1e-8f);
        wave[idx] = tanhf(wave[idx] / wm);
    }
}

// ---------------------------------------------------------------------------
extern "C" void kernel_launch(void* const* d_in, const int* in_sizes, int n_in,
                              void* d_out, int out_size, void* d_ws, size_t ws_size,
                              hipStream_t stream)
{
    const float* mel = (const float*)d_in[0];
    const float* f0  = (const float*)d_in[1];
    const float* voi = (const float*)d_in[2];
    const float* w_f1 = (const float*)d_in[3];  const float* b_f1 = (const float*)d_in[4];
    const float* w_f2 = (const float*)d_in[5];  const float* b_f2 = (const float*)d_in[6];
    const float* w_a1 = (const float*)d_in[7];  const float* b_a1 = (const float*)d_in[8];
    const float* g1 = (const float*)d_in[9];    const float* be1 = (const float*)d_in[10];
    const float* m1 = (const float*)d_in[11];   const float* v1  = (const float*)d_in[12];
    const float* w_a2 = (const float*)d_in[13]; const float* b_a2 = (const float*)d_in[14];
    const float* g2 = (const float*)d_in[15];   const float* be2 = (const float*)d_in[16];
    const float* m2 = (const float*)d_in[17];   const float* v2  = (const float*)d_in[18];
    const float* w_a3 = (const float*)d_in[19]; const float* b_a3 = (const float*)d_in[20];
    const float* w_p1 = (const float*)d_in[21]; const float* b_p1 = (const float*)d_in[22];
    const float* w_p2 = (const float*)d_in[23]; const float* b_p2 = (const float*)d_in[24];

    char* wsp = (char*)d_ws;
    size_t off = 0;
    auto alloc = [&](size_t bytes) -> char* {
        char* p = wsp + off;
        off = (off + bytes + 255) & ~(size_t)255;
        return p;
    };
    _Float16* comb   = (_Float16*)alloc((size_t)BB * 96 * TF * 2);   // mel + f0_feat
    _Float16* act1   = (_Float16*)alloc((size_t)BB * 128 * TF * 2);
    _Float16* act2   = (_Float16*)alloc((size_t)BB * 128 * TF * 2);
    _Float16* actp1  = (_Float16*)alloc((size_t)BB * 64 * TF * 2);
    _Float16* w16a1  = (_Float16*)alloc((size_t)128 * 256 * 2);  // Kpad(240)=256
    _Float16* w16a2  = (_Float16*)alloc((size_t)128 * 384 * 2);
    _Float16* w16a3  = (_Float16*)alloc((size_t)32 * 384 * 2);
    _Float16* w16p1  = (_Float16*)alloc((size_t)64 * 288 * 2);
    _Float16* w16p2  = (_Float16*)alloc((size_t)32 * 192 * 2);
    float* ba1 = (float*)alloc(128 * 4);
    float* ba2 = (float*)alloc(128 * 4);
    float* ba3 = (float*)alloc(32 * 4);
    float* bp1 = (float*)alloc(64 * 4);
    float* bp2 = (float*)alloc(32 * 4);
    float* f0hz   = (float*)alloc((size_t)BB * TF * 4);
    float* amps   = (float*)alloc((size_t)BB * HH * TF * 4);
    float* phases = (float*)alloc((size_t)BB * HH * TF * 4);
    float* pbase  = (float*)alloc((size_t)BB * HH * TF * 4);
    float* pinc   = (float*)alloc((size_t)BB * HH * TF * 4);
    unsigned* wmaxBits = (unsigned*)alloc(BB * 4);
    float* out = (float*)d_out;

    // --- prep ---
    f0_process<<<1, 256, 0, stream>>>(f0, voi, f0hz, wmaxBits);
    mel_to_comb<<<1000, 256, 0, stream>>>(mel, comb);
    f0_branch<<<BB, 256, 0, stream>>>(f0, w_f1, b_f1, w_f2, b_f2, comb);
    fold_weights<<<128, 256, 0, stream>>>(w_a1, b_a1, g1, be1, m1, v1, w16a1, ba1, 128, 240, 256);
    fold_weights<<<192, 256, 0, stream>>>(w_a2, b_a2, g2, be2, m2, v2, w16a2, ba2, 128, 384, 384);
    fold_weights<<<48, 256, 0, stream>>>(w_a3, b_a3, nullptr, nullptr, nullptr, nullptr, w16a3, ba3, 32, 384, 384);
    fold_weights<<<72, 256, 0, stream>>>(w_p1, b_p1, nullptr, nullptr, nullptr, nullptr, w16p1, bp1, 64, 288, 288);
    fold_weights<<<24, 256, 0, stream>>>(w_p2, b_p2, nullptr, nullptr, nullptr, nullptr, w16p2, bp2, 32, 192, 192);

    // --- WMMA conv stacks (TDM-staged X tiles) ---
    conv_wmma<80, 128, 0><<<dim3(2, 25, BB), 128, 0, stream>>>(comb,  w16a1, ba1, act1,  nullptr, nullptr, 96);
    conv_wmma<128, 128, 0><<<dim3(2, 25, BB), 128, 0, stream>>>(act1, w16a2, ba2, act2,  nullptr, nullptr, 128);
    conv_wmma<128, 32, 1><<<dim3(1, 25, BB), 64, 0, stream>>>(act2,  w16a3, ba3, nullptr, amps,  voi,     128);
    conv_wmma<96, 64, 0><<<dim3(1, 25, BB), 128, 0, stream>>>(comb,  w16p1, bp1, actp1, nullptr, nullptr, 96);
    conv_wmma<64, 32, 2><<<dim3(1, 25, BB), 64, 0, stream>>>(actp1, w16p2, bp2, nullptr, phases, nullptr, 64);

    // --- harmonic synthesis ---
    synth_scan<<<BB, 32, 0, stream>>>(f0hz, phases, pbase, pinc);
    synth_wave<<<dim3(BB, 25), 256, 0, stream>>>(pbase, pinc, amps, out, wmaxBits);
    normalize_out<<<3000, 256, 0, stream>>>(out, wmaxBits);
}
